// SemmaRelModel_68908455297612
// MI455X (gfx1250) — matrix operand order repaired and checked by reading, not compiled, not run on hardware
//
#include <hip/hip_runtime.h>
#include <hip/hip_bf16.h>

typedef __attribute__((ext_vector_type(16))) __bf16 v16bf;
typedef __attribute__((ext_vector_type(8)))  __bf16 v8bf;
typedef __attribute__((ext_vector_type(8)))  float  v8f;
typedef __attribute__((ext_vector_type(4)))  float  v4f;

#define D 64
#define KDIM 128

// ---------- helpers ----------

__device__ inline v16bf load16(const __bf16* p) {
    const v8bf* q = (const v8bf*)p;   // 32B contiguous, 16B aligned
    v8bf lo = q[0], hi = q[1];
    v16bf r;
#pragma unroll
    for (int i = 0; i < 8; ++i) { r[i] = lo[i]; r[i + 8] = hi[i]; }
    return r;
}

// ---------- tiny elementwise kernels ----------

__global__ void zero_f4(float* __restrict__ p, int n4) {
    int t = blockIdx.x * 256 + threadIdx.x;
    if (t < n4) ((v4f*)p)[t] = (v4f){0.f, 0.f, 0.f, 0.f};
}

// dst[b][n][d] = src[n][d]  (broadcast over batch)
__global__ void bcast_f4(float* __restrict__ dst, const float* __restrict__ src,
                         int n4_total, int n4_per_batch) {
    int t = blockIdx.x * 256 + threadIdx.x;
    if (t < n4_total) ((v4f*)dst)[t] = ((const v4f*)src)[t % n4_per_batch];
}

// x[b][h_index[b]][d] = 1.0  (buffer pre-zeroed)
__global__ void indicator_kernel(float* __restrict__ x, const int* __restrict__ hidx,
                                 int B, int N) {
    int t = blockIdx.x * 256 + threadIdx.x;
    if (t >= B * D) return;
    int b = t / D, d = t % D;
    x[((long)b * N + hidx[b]) * D + d] = 1.0f;
}

// pack W[K][64] f32 -> bf16 WMMA B-fragments: dst[((kt*4+nt)*32+L)*16+i]
__global__ void pack_w(const float* __restrict__ W, __bf16* __restrict__ dst, int K) {
    int t = blockIdx.x * 256 + threadIdx.x;
    if (t >= K * 64) return;
    int i  = t & 15;
    int L  = (t >> 4) & 31;
    int nt = (t >> 9) & 3;
    int kt = t >> 11;
    int krow = kt * 32 + ((L >> 4) << 4) + i;   // K = kt*32 + 16*(L>=16) + i
    int col  = nt * 16 + (L & 15);
    dst[t] = (__bf16)W[krow * 64 + col];
}

// ---------- edge message + scatter-add ----------
// one 32-lane slice per edge, 2 floats/lane (coalesced 256B row access, L2 resident)
__global__ void scatter_kernel(const float* __restrict__ x, float* __restrict__ agg,
                               const int* __restrict__ src, const int* __restrict__ dst,
                               const int* __restrict__ etype, const float* __restrict__ rel,
                               int nEdges, int B, int N) {
    long t = (long)blockIdx.x * 256 + threadIdx.x;
    long e = t >> 5;
    int  c = (int)(t & 31) * 2;
    if (e >= nEdges) return;
    int s = src[e], d2 = dst[e];
    int ty = etype ? etype[e] : 0;
    float rv0 = rel[ty * D + c], rv1 = rel[ty * D + c + 1];
    for (int b = 0; b < B; ++b) {
        long sb = ((long)b * N + s) * D + c;
        long db = ((long)b * N + d2) * D + c;
        atomicAdd(agg + db,     x[sb]     * rv0);
        atomicAdd(agg + db + 1, x[sb + 1] * rv1);
    }
}

// ---------- fused layer: h = relu(LN(concat[x,agg] @ W + b)) + x  (in-place x) ----------
// 8 waves/block, 16 rows/wave, WMMA 16x16x32 bf16
__global__ __launch_bounds__(256) void layer_kernel(
    float* __restrict__ x, const float* __restrict__ agg,
    const __bf16* __restrict__ wpk, const float* __restrict__ bias,
    const float* __restrict__ gamma, const float* __restrict__ beta, int nRows) {
    __shared__ __attribute__((aligned(16))) __bf16 smem[8][16][KDIM];

    int lane = threadIdx.x & 31;
    int wave = threadIdx.x >> 5;
    long rowBase = (long)blockIdx.x * 128 + (long)wave * 16;
    __bf16* At = &smem[wave][0][0];

    // stage 16 rows x 128 cols (cols 0-63 from x, 64-127 from agg) as bf16
    int scol = (lane & 15) * 4;
    const float* sbase = (lane < 16) ? x : agg;
#pragma unroll 4
    for (int r = 0; r < 16; ++r) {
        long row = rowBase + r;
        v4f v = (v4f){0.f, 0.f, 0.f, 0.f};
        if (row < nRows) v = *(const v4f*)(sbase + row * D + scol);
        __bf16* dp = At + r * KDIM + ((lane < 16) ? scol : (64 + scol));
        dp[0] = (__bf16)v[0]; dp[1] = (__bf16)v[1];
        dp[2] = (__bf16)v[2]; dp[3] = (__bf16)v[3];
    }
    __syncthreads();

    int mrow = lane & 15;
    int half = lane >> 4;
    v8f acc[4];
#pragma unroll
    for (int nt = 0; nt < 4; ++nt) acc[nt] = (v8f){0.f,0.f,0.f,0.f,0.f,0.f,0.f,0.f};

#pragma unroll
    for (int kt = 0; kt < 4; ++kt) {
        v16bf a = load16(At + mrow * KDIM + kt * 32 + half * 16);
#pragma unroll
        for (int nt = 0; nt < 4; ++nt) {
            v16bf bb = load16(wpk + ((kt * 4 + nt) * 32 + lane) * 16);
            acc[nt] = __builtin_amdgcn_wmma_f32_16x16x32_bf16(
                false, a, false, bb, (short)0, acc[nt], false, false);
        }
    }
    __syncthreads();

    // epilogue: acc -> LDS f32 tile (reuse A region), + bias
    float* Ot = (float*)At;  // 16 x 64 f32 = 4KB, same 4KB as A tile
    int col = lane & 15;
#pragma unroll
    for (int nt = 0; nt < 4; ++nt) {
        float bv = bias[nt * 16 + col];
#pragma unroll
        for (int r = 0; r < 8; ++r)
            Ot[(r + 8 * half) * D + nt * 16 + col] = acc[nt][r] + bv;
    }
    __syncthreads();

    // per-row LayerNorm + relu + residual (lane<16: one row each)
    if (lane < 16) {
        long row = rowBase + lane;
        if (row < nRows) {
            const float* rp = Ot + lane * D;
            float s = 0.f;
#pragma unroll
            for (int c = 0; c < D; ++c) s += rp[c];
            float mean = s * (1.f / D);
            float s2 = 0.f;
#pragma unroll
            for (int c = 0; c < D; ++c) { float dd = rp[c] - mean; s2 += dd * dd; }
            float inv = rsqrtf(s2 * (1.f / D) + 1e-5f);
#pragma unroll
            for (int c = 0; c < D; ++c) {
                float hn = (rp[c] - mean) * inv * gamma[c] + beta[c];
                hn = fmaxf(hn, 0.f);
                x[row * D + c] = hn + x[row * D + c];
            }
        }
    }
}

// ---------- fusion: out = relu(concat[h,z] @ W1 + b1) @ W2 + b2 ----------
__global__ __launch_bounds__(256) void fuse_kernel(
    const float* __restrict__ h, const float* __restrict__ z,
    const __bf16* __restrict__ w1pk, const float* __restrict__ b1,
    const __bf16* __restrict__ w2pk, const float* __restrict__ b2,
    float* __restrict__ out, int nRows) {
    __shared__ __attribute__((aligned(16))) __bf16 smem[8][16][KDIM];

    int lane = threadIdx.x & 31;
    int wave = threadIdx.x >> 5;
    long rowBase = (long)blockIdx.x * 128 + (long)wave * 16;
    __bf16* At = &smem[wave][0][0];

    int scol = (lane & 15) * 4;
    const float* sbase = (lane < 16) ? h : z;
#pragma unroll 4
    for (int r = 0; r < 16; ++r) {
        long row = rowBase + r;
        v4f v = (v4f){0.f, 0.f, 0.f, 0.f};
        if (row < nRows) v = *(const v4f*)(sbase + row * D + scol);
        __bf16* dp = At + r * KDIM + ((lane < 16) ? scol : (64 + scol));
        dp[0] = (__bf16)v[0]; dp[1] = (__bf16)v[1];
        dp[2] = (__bf16)v[2]; dp[3] = (__bf16)v[3];
    }
    __syncthreads();

    int mrow = lane & 15;
    int half = lane >> 4;
    v8f acc[4];
#pragma unroll
    for (int nt = 0; nt < 4; ++nt) acc[nt] = (v8f){0.f,0.f,0.f,0.f,0.f,0.f,0.f,0.f};

#pragma unroll
    for (int kt = 0; kt < 4; ++kt) {
        v16bf a = load16(At + mrow * KDIM + kt * 32 + half * 16);
#pragma unroll
        for (int nt = 0; nt < 4; ++nt) {
            v16bf bb = load16(w1pk + ((kt * 4 + nt) * 32 + lane) * 16);
            acc[nt] = __builtin_amdgcn_wmma_f32_16x16x32_bf16(
                false, a, false, bb, (short)0, acc[nt], false, false);
        }
    }
    __syncthreads();

    // relu(acc + b1) -> bf16 tile (16 x 64, row stride 64) for GEMM2 A-matrix
    __bf16* Ft = At;
    int col = lane & 15;
#pragma unroll
    for (int nt = 0; nt < 4; ++nt) {
        float bv = b1[nt * 16 + col];
#pragma unroll
        for (int r = 0; r < 8; ++r)
            Ft[(r + 8 * half) * D + nt * 16 + col] = (__bf16)fmaxf(acc[nt][r] + bv, 0.f);
    }
    __syncthreads();

    v8f acc2[4];
#pragma unroll
    for (int nt = 0; nt < 4; ++nt) acc2[nt] = (v8f){0.f,0.f,0.f,0.f,0.f,0.f,0.f,0.f};
#pragma unroll
    for (int kt = 0; kt < 2; ++kt) {
        v16bf a = load16(Ft + mrow * D + kt * 32 + half * 16);
#pragma unroll
        for (int nt = 0; nt < 4; ++nt) {
            v16bf bb = load16(w2pk + ((kt * 4 + nt) * 32 + lane) * 16);
            acc2[nt] = __builtin_amdgcn_wmma_f32_16x16x32_bf16(
                false, a, false, bb, (short)0, acc2[nt], false, false);
        }
    }

#pragma unroll
    for (int nt = 0; nt < 4; ++nt) {
        float bv = b2[nt * 16 + col];
#pragma unroll
        for (int r = 0; r < 8; ++r) {
            long row = rowBase + r + 8 * half;
            if (row < nRows) out[row * D + nt * 16 + col] = acc2[nt][r] + bv;
        }
    }
}

// ---------- host ----------

extern "C" void kernel_launch(void* const* d_in, const int* in_sizes, int n_in,
                              void* d_out, int out_size, void* d_ws, size_t ws_size,
                              hipStream_t stream) {
    const float* rel_text_init = (const float*)d_in[0];
    const float* struct_rel    = (const float*)d_in[1];
    const float* struct_W      = (const float*)d_in[2];
    const float* struct_b      = (const float*)d_in[3];
    const float* struct_g      = (const float*)d_in[4];
    const float* struct_beta   = (const float*)d_in[5];
    const float* text_rel      = (const float*)d_in[6];
    const float* text_W        = (const float*)d_in[7];
    const float* text_b        = (const float*)d_in[8];
    const float* text_g        = (const float*)d_in[9];
    const float* text_beta     = (const float*)d_in[10];
    const float* fuse_W1       = (const float*)d_in[11];
    const float* fuse_b1       = (const float*)d_in[12];
    const float* fuse_W2       = (const float*)d_in[13];
    const float* fuse_b2       = (const float*)d_in[14];
    const int*   h_index       = (const int*)d_in[15];
    const int*   edge_index    = (const int*)d_in[16];
    const int*   edge_type     = (const int*)d_in[17];
    const int*   text_edge     = (const int*)d_in[18];

    const int N  = in_sizes[0] / D;          // 50000
    const int B  = in_sizes[15];             // 2
    const int L  = in_sizes[3] / D;          // 6
    const int E  = in_sizes[16] / 2;         // 400000
    const int ET = in_sizes[18] / 2;         // 200000
    const int nRows = B * N;                 // 100000
    const int TOT   = nRows * D;             // 6.4M floats

    float*  xs  = (float*)d_ws;
    float*  xt  = xs + TOT;
    float*  agg = xt + TOT;
    __bf16* wpk = (__bf16*)(agg + TOT);
    __bf16* wpkS  = wpk;                      // L * 8192
    __bf16* wpkT  = wpkS + (size_t)L * KDIM * D;
    __bf16* wpkW1 = wpkT + (size_t)L * KDIM * D;
    __bf16* wpkW2 = wpkW1 + KDIM * D;

    const int BS = 256;
    dim3 blk(BS);
    int n4      = TOT / 4;
    dim3 gZero((n4 + BS - 1) / BS);
    dim3 gBcast(gZero);
    dim3 gInd((B * D + BS - 1) / BS);
    dim3 gPack128((KDIM * D + BS - 1) / BS);
    dim3 gPack64((D * D + BS - 1) / BS);
    dim3 gGemm((nRows + 127) / 128);
    dim3 gScatS((int)(((long)E * 32 + BS - 1) / BS));
    dim3 gScatT((int)(((long)ET * 32 + BS - 1) / BS));

    // pack all weights to bf16 WMMA-fragment layout
    for (int l = 0; l < L; ++l) {
        pack_w<<<gPack128, blk, 0, stream>>>(struct_W + (size_t)l * KDIM * D,
                                             wpkS + (size_t)l * KDIM * D, KDIM);
        pack_w<<<gPack128, blk, 0, stream>>>(text_W + (size_t)l * KDIM * D,
                                             wpkT + (size_t)l * KDIM * D, KDIM);
    }
    pack_w<<<gPack128, blk, 0, stream>>>(fuse_W1, wpkW1, KDIM);
    pack_w<<<gPack64,  blk, 0, stream>>>(fuse_W2, wpkW2, D);

    // ---- struct branch: boundary = indicator at h_index ----
    zero_f4<<<gZero, blk, 0, stream>>>(xs, n4);
    indicator_kernel<<<gInd, blk, 0, stream>>>(xs, h_index, B, N);
    for (int l = 0; l < L; ++l) {
        zero_f4<<<gZero, blk, 0, stream>>>(agg, n4);
        indicator_kernel<<<gInd, blk, 0, stream>>>(agg, h_index, B, N);
        scatter_kernel<<<gScatS, blk, 0, stream>>>(
            xs, agg, edge_index, edge_index + E, edge_type,
            struct_rel + (size_t)l * 4 * D, E, B, N);
        layer_kernel<<<gGemm, blk, 0, stream>>>(
            xs, agg, wpkS + (size_t)l * KDIM * D,
            struct_b + (size_t)l * D, struct_g + (size_t)l * D,
            struct_beta + (size_t)l * D, nRows);
    }

    // ---- text branch: boundary = broadcast(rel_text_init) ----
    bcast_f4<<<gBcast, blk, 0, stream>>>(xt, rel_text_init, n4, (N * D) / 4);
    for (int l = 0; l < L; ++l) {
        bcast_f4<<<gBcast, blk, 0, stream>>>(agg, rel_text_init, n4, (N * D) / 4);
        scatter_kernel<<<gScatT, blk, 0, stream>>>(
            xt, agg, text_edge, text_edge + ET, (const int*)nullptr,
            text_rel + (size_t)l * D, ET, B, N);
        layer_kernel<<<gGemm, blk, 0, stream>>>(
            xt, agg, wpkT + (size_t)l * KDIM * D,
            text_b + (size_t)l * D, text_g + (size_t)l * D,
            text_beta + (size_t)l * D, nRows);
    }

    // ---- fusion MLP ----
    fuse_kernel<<<gGemm, blk, 0, stream>>>(xs, xt, wpkW1, fuse_b1,
                                           wpkW2, fuse_b2, (float*)d_out, nRows);
}
